// IntraClusterGAT_1666447311292
// MI455X (gfx1250) — matrix operand order repaired and checked by reading, not compiled, not run on hardware
//
#include <hip/hip_runtime.h>
#include <hip/hip_bf16.h>
#include <cstdint>

typedef __attribute__((ext_vector_type(16))) _Float16 v16h;
typedef __attribute__((ext_vector_type(8)))  float    v8f;

union Frag16 { v16h h; uint4 u[2]; };

#define NUM_VARS     100000
#define NUM_NODES    200000
#define NUM_CLUSTERS 2048
#define DIM          64
#define NEG_SLOPE    0.2f
#define GAMMA        1.0f

static __device__ __forceinline__ v8f wmma16(const v16h& a, const v16h& b, const v8f& c) {
  // D = A(16x32 f16) * B(32x16 f16) + C(16x16 f32)
  return __builtin_amdgcn_wmma_f32_16x16x32_f16(false, a, false, b, (short)0, c, false, false);
}

// gfx1250 async global->LDS copy (ASYNCcnt-tracked). LDS operand is the
// 32-bit LDS byte offset = low 32 bits of the generic LDS pointer.
static __device__ __forceinline__ void async_ld16(void* lds_dst, const void* gsrc) {
  uint32_t loff = (uint32_t)(uintptr_t)lds_dst;
  unsigned long long ga = (unsigned long long)(uintptr_t)gsrc;
  asm volatile("global_load_async_to_lds_b128 %0, %1, off"
               :: "v"(loff), "v"(ga) : "memory");
}
static __device__ __forceinline__ void wait_async0() {
  asm volatile("s_wait_asynccnt 0x0" ::: "memory");
}

// ---------------------------------------------------------------------------
// Kernel A: Q/K/V projection.  x_all[200000x64] f32 -> Qh/Kh/Vh f16.
// Wave handles 16 rows; W^T preloaded to LDS as f16 (rows padded: 72 halves).
// ---------------------------------------------------------------------------
__global__ __launch_bounds__(256)
void qkv_kernel(const float* __restrict__ xv, const float* __restrict__ xc,
                const float* __restrict__ WQ, const float* __restrict__ WK,
                const float* __restrict__ WV,
                _Float16* __restrict__ Qh, _Float16* __restrict__ Kh,
                _Float16* __restrict__ Vh, int num_tiles) {
  __shared__ __align__(16) _Float16 wt[3][64][72];   // wt[mat][d][j] = W[mat][j][d]
  int tid = threadIdx.x;
  for (int idx = tid; idx < 3 * 64 * 64; idx += 256) {
    int mat = idx >> 12; int r = idx & 4095; int j = r >> 6; int d = r & 63;
    const float* W = (mat == 0) ? WQ : ((mat == 1) ? WK : WV);
    wt[mat][d][j] = (_Float16)W[j * 64 + d];
  }
  __syncthreads();

  int wave = tid >> 5, lane = tid & 31;
  int tile = blockIdx.x * 8 + wave;
  if (tile >= num_tiles) return;
  int m = lane & 15, hi = lane >> 4;

  int row = tile * 16 + m;
  const float* xrow = (row < NUM_VARS) ? (xv + (size_t)row * DIM)
                                       : (xc + (size_t)(row - NUM_VARS) * DIM);
  // A fragments (K chunks 0..31, 32..63), ISA 16-bit A layout
  Frag16 a[2];
#pragma unroll
  for (int c = 0; c < 2; ++c) {
    int s0 = c * 32 + hi * 8;
    const float* p0 = xrow + s0;
    const float* p1 = xrow + s0 + 16;
#pragma unroll
    for (int e = 0; e < 8; ++e) {
      a[c].h[e]     = (_Float16)p0[e];
      a[c].h[e + 8] = (_Float16)p1[e];
    }
  }

#pragma unroll
  for (int mat = 0; mat < 3; ++mat) {
    _Float16* outp = (mat == 0) ? Qh : ((mat == 1) ? Kh : Vh);
#pragma unroll
    for (int jt = 0; jt < 4; ++jt) {
      v8f acc = {};
#pragma unroll
      for (int c = 0; c < 2; ++c) {
        Frag16 b;                               // lane = N-col of B, elems = K
#pragma unroll
        for (int e = 0; e < 8; ++e) {
          b.h[e]     = wt[mat][c * 32 + hi * 8 + e][jt * 16 + m];
          b.h[e + 8] = wt[mat][c * 32 + 16 + hi * 8 + e][jt * 16 + m];
        }
        acc = wmma16(a[c].h, b.h, acc);
      }
#pragma unroll
      for (int r = 0; r < 8; ++r) {             // D layout: M = r + hi*8, N = m
        int rr = tile * 16 + r + hi * 8;
        outp[(size_t)rr * DIM + jt * 16 + m] = (_Float16)acc[r];
      }
    }
  }
}

// ---------------------------------------------------------------------------
// Kernel B: per-cluster attention (block = cluster, 8 waves, wave = 16 rows).
// LDS rows padded to decluster banks; Q/K region overlaid by attn buffer.
// ---------------------------------------------------------------------------
__global__ __launch_bounds__(256)
void attn_kernel(const _Float16* __restrict__ Qh, const _Float16* __restrict__ Kh,
                 const _Float16* __restrict__ Vh,
                 const int* __restrict__ var_ids, const int* __restrict__ clause_ids,
                 const float* __restrict__ sat,
                 const int* __restrict__ active_heads, const float* __restrict__ head_w,
                 float* __restrict__ accum, float* __restrict__ count) {
  __shared__ __align__(16) union {
    struct { _Float16 q[128][72]; _Float16 kt[64][136]; } qk;  // 35840 B
    _Float16 attn[8][16][136];                                  // 34816 B
  } u;
  __shared__ __align__(16) _Float16 vsm[128][72];
  __shared__ float bias[128];
  __shared__ int   nodes[128];

  int c = blockIdx.x;
  int tid = threadIdx.x;

  if (tid < 128) {
    int nd; float b;
    if (tid < 64) { nd = var_ids[c * 64 + tid]; b = 0.f; }
    else { int cl = clause_ids[c * 64 + tid - 64]; nd = NUM_VARS + cl; b = GAMMA * sat[cl]; }
    nodes[tid] = nd; bias[tid] = b;
  }
  __syncthreads();

  { // gather: 2 threads per slot, 32 halves (64B) each.
    int slot = tid >> 1, h = tid & 1;
    int nd = nodes[slot];
    const _Float16* qsrc = Qh + (size_t)nd * DIM + h * 32;
    const _Float16* ksrc = Kh + (size_t)nd * DIM + h * 32;
    const _Float16* vsrc = Vh + (size_t)nd * DIM + h * 32;
    // Q, V: row-major consumers -> async global->LDS (no VGPR staging)
#pragma unroll
    for (int i = 0; i < 4; ++i) {
      async_ld16(&u.qk.q[slot][h * 32 + i * 8], qsrc + i * 8);
      async_ld16(&vsm[slot][h * 32 + i * 8],    vsrc + i * 8);
    }
    // K: needs transpose -> VGPR round trip
    _Float16 kr[32];
#pragma unroll
    for (int i = 0; i < 4; ++i) *(uint4*)&kr[i * 8] = *(const uint4*)(ksrc + i * 8);
#pragma unroll
    for (int d = 0; d < 32; ++d) u.qk.kt[h * 32 + d][slot] = kr[d];
  }
  wait_async0();
  __syncthreads();

  int wave = tid >> 5, lane = tid & 31;
  int m = lane & 15, hi = lane >> 4;

  Frag16 qa[2];
#pragma unroll
  for (int ck = 0; ck < 2; ++ck) {
    const _Float16* p = &u.qk.q[wave * 16 + m][ck * 32 + hi * 8];
    qa[ck].u[0] = *(const uint4*)p;
    qa[ck].u[1] = *(const uint4*)(p + 16);
  }

  v8f sc[8];                                    // 16x128 scores in registers
#pragma unroll
  for (int nt = 0; nt < 8; ++nt) {
    v8f acc = {};
#pragma unroll
    for (int ck = 0; ck < 2; ++ck) {
      Frag16 b;
#pragma unroll
      for (int e = 0; e < 8; ++e) {
        b.h[e]     = u.qk.kt[ck * 32 + hi * 8 + e][nt * 16 + m];
        b.h[e + 8] = u.qk.kt[ck * 32 + 16 + hi * 8 + e][nt * 16 + m];
      }
      acc = wmma16(qa[ck].h, b.h, acc);
    }
    float bn = bias[nt * 16 + m];               // column bias (N = m)
#pragma unroll
    for (int r = 0; r < 8; ++r) {
      float v = acc[r] * 0.125f + bn;           // /sqrt(64)
      acc[r] = (v > 0.f) ? v : v * NEG_SLOPE;   // leaky_relu
    }
    sc[nt] = acc;
  }

  // row softmax: row M = r + hi*8 lives in one 16-lane half -> xor masks 1..8
  float rinv[8];
#pragma unroll
  for (int r = 0; r < 8; ++r) {
    float mx = -3.0e38f;
#pragma unroll
    for (int nt = 0; nt < 8; ++nt) mx = fmaxf(mx, sc[nt][r]);
#pragma unroll
    for (int k = 8; k >= 1; k >>= 1) mx = fmaxf(mx, __shfl_xor(mx, k, 32));
    float sm = 0.f;
#pragma unroll
    for (int nt = 0; nt < 8; ++nt) { float e = __expf(sc[nt][r] - mx); sc[nt][r] = e; sm += e; }
#pragma unroll
    for (int k = 8; k >= 1; k >>= 1) sm += __shfl_xor(sm, k, 32);
    rinv[r] = 1.0f / sm;
  }

  __syncthreads();                              // q/kt dead -> overlay attn
#pragma unroll
  for (int nt = 0; nt < 8; ++nt)
#pragma unroll
    for (int r = 0; r < 8; ++r)
      u.attn[wave][r + hi * 8][nt * 16 + m] = (_Float16)(sc[nt][r] * rinv[r]);
  // wave reads only its own attn region; same-wave LDS ops are in order

  int ah = active_heads[0]; if (ah < 1) ah = 1; if (ah > 4) ah = 4;
  float hw = 0.f;
  for (int i = 0; i < ah; ++i) hw += head_w[i];
  hw /= (float)ah;

  Frag16 af[4];                                 // attn A-frags: invariant over jt
#pragma unroll
  for (int kt = 0; kt < 4; ++kt) {
    const _Float16* pa = &u.attn[wave][m][kt * 32 + hi * 8];
    af[kt].u[0] = *(const uint4*)pa;
    af[kt].u[1] = *(const uint4*)(pa + 16);
  }

#pragma unroll
  for (int jt = 0; jt < 4; ++jt) {
    v8f acc = {};
#pragma unroll
    for (int kt = 0; kt < 4; ++kt) {            // K = 128 -> 4 x k32
      Frag16 b;
#pragma unroll
      for (int e = 0; e < 8; ++e) {
        b.h[e]     = vsm[kt * 32 + hi * 8 + e][jt * 16 + m];
        b.h[e + 8] = vsm[kt * 32 + 16 + hi * 8 + e][jt * 16 + m];
      }
      acc = wmma16(af[kt].h, b.h, acc);
    }
#pragma unroll
    for (int r = 0; r < 8; ++r) {
      int nd = nodes[wave * 16 + r + hi * 8];
      atomicAdd(&accum[(size_t)nd * DIM + jt * 16 + m], acc[r] * hw);
    }
  }
  if (lane < 16) atomicAdd(&count[nodes[wave * 16 + lane]], 1.0f);
}

// ---------------------------------------------------------------------------
// Kernel C: mean-normalize, out-projection (WMMA), bias + residual -> d_out.
// ---------------------------------------------------------------------------
__global__ __launch_bounds__(256)
void proj_kernel(const float* __restrict__ accum, const float* __restrict__ count,
                 const float* __restrict__ xv, const float* __restrict__ xc,
                 const float* __restrict__ Wo, const float* __restrict__ bo,
                 float* __restrict__ out, int num_tiles) {
  __shared__ __align__(16) _Float16 wt[64][72];      // wt[d][j] = Wo[j][d]
  int tid = threadIdx.x;
  for (int idx = tid; idx < 4096; idx += 256) {
    int j = idx >> 6, d = idx & 63;
    wt[d][j] = (_Float16)Wo[j * 64 + d];
  }
  __syncthreads();

  int wave = tid >> 5, lane = tid & 31;
  int tile = blockIdx.x * 8 + wave;
  if (tile >= num_tiles) return;
  int m = lane & 15, hi = lane >> 4;

  int row = tile * 16 + m;
  float inv = 1.0f / fmaxf(count[row], 1.0f);
  const float* arow = accum + (size_t)row * DIM;
  Frag16 a[2];
#pragma unroll
  for (int c = 0; c < 2; ++c) {
    int s0 = c * 32 + hi * 8;
#pragma unroll
    for (int e = 0; e < 8; ++e) {
      a[c].h[e]     = (_Float16)(arow[s0 + e] * inv);
      a[c].h[e + 8] = (_Float16)(arow[s0 + 16 + e] * inv);
    }
  }

#pragma unroll
  for (int jt = 0; jt < 4; ++jt) {
    v8f acc = {};
#pragma unroll
    for (int c = 0; c < 2; ++c) {
      Frag16 b;
#pragma unroll
      for (int e = 0; e < 8; ++e) {
        b.h[e]     = wt[c * 32 + hi * 8 + e][jt * 16 + m];
        b.h[e + 8] = wt[c * 32 + 16 + hi * 8 + e][jt * 16 + m];
      }
      acc = wmma16(a[c].h, b.h, acc);
    }
    float bj = bo[jt * 16 + m];
#pragma unroll
    for (int r = 0; r < 8; ++r) {
      int rr = tile * 16 + r + hi * 8;
      const float* xrow = (rr < NUM_VARS) ? (xv + (size_t)rr * DIM)
                                          : (xc + (size_t)(rr - NUM_VARS) * DIM);
      out[(size_t)rr * DIM + jt * 16 + m] = xrow[jt * 16 + m] + acc[r] + bj;
    }
  }
}

// ---------------------------------------------------------------------------
extern "C" void kernel_launch(void* const* d_in, const int* in_sizes, int n_in,
                              void* d_out, int out_size, void* d_ws, size_t ws_size,
                              hipStream_t stream) {
  (void)in_sizes; (void)n_in; (void)out_size; (void)ws_size;
  const float* xv         = (const float*)d_in[0];
  const float* xc         = (const float*)d_in[1];
  // d_in[2] edge_index, d_in[3] edge_polarity: unused by reference
  const int*   var_ids    = (const int*)d_in[4];
  const int*   clause_ids = (const int*)d_in[5];
  const float* sat        = (const float*)d_in[6];
  const int*   active_h   = (const int*)d_in[7];
  const float* WQ         = (const float*)d_in[8];
  const float* WK         = (const float*)d_in[9];
  const float* WV         = (const float*)d_in[10];
  const float* headw      = (const float*)d_in[11];
  const float* Wo         = (const float*)d_in[12];
  const float* bo         = (const float*)d_in[13];

  char* ws = (char*)d_ws;
  const size_t sz_h = (size_t)NUM_NODES * DIM * sizeof(_Float16);   // 25.6 MB
  _Float16* Qh    = (_Float16*)(ws);
  _Float16* Kh    = (_Float16*)(ws + sz_h);
  _Float16* Vh    = (_Float16*)(ws + 2 * sz_h);
  float*    accum = (float*)(ws + 3 * sz_h);
  float*    count = (float*)(ws + 3 * sz_h + (size_t)NUM_NODES * DIM * sizeof(float));

  // zero accumulators (accum + count are adjacent)
  hipMemsetAsync(accum, 0,
                 (size_t)NUM_NODES * DIM * sizeof(float) + (size_t)NUM_NODES * sizeof(float),
                 stream);

  int num_tiles = NUM_NODES / 16;               // 12500
  int blocksA = (num_tiles + 7) / 8;

  qkv_kernel<<<blocksA, 256, 0, stream>>>(xv, xc, WQ, WK, WV, Qh, Kh, Vh, num_tiles);
  attn_kernel<<<NUM_CLUSTERS, 256, 0, stream>>>(Qh, Kh, Vh, var_ids, clause_ids, sat,
                                                active_h, headw, accum, count);
  proj_kernel<<<blocksA, 256, 0, stream>>>(accum, count, xv, xc, Wo, bo,
                                           (float*)d_out, num_tiles);
}
